// ViTSelfAttention_3350074491120
// MI455X (gfx1250) — matrix operand order repaired and checked by reading, not compiled, not run on hardware
//
#include <hip/hip_runtime.h>
#include <hip/hip_bf16.h>

// ViT self-attention forward on gfx1250 (MI455X).
// GEMMs: v_wmma_f32_16x16x32_f16 with block-shared B panels staged in LDS via
// global_load_async_to_lds_b128 (double-buffered, ASYNCcnt-decoupled).
// Attention: flash-style, async double-buffered K/V chunks in LDS.
// All LDS tiles padded to conflict-free bank strides.

typedef _Float16 half_t;
typedef __attribute__((ext_vector_type(16))) _Float16 v16h;
typedef __attribute__((ext_vector_type(8)))  _Float16 v8h;
typedef __attribute__((ext_vector_type(8)))  float    v8f;
typedef __attribute__((ext_vector_type(4)))  int      v4i;

#define S_LEN   1024
#define HID     1024
#define NH      16
#define HD      64
#define BATCH   8
#define TOKENS  (BATCH * S_LEN)   // 8192

#define BSTR    40   // LDS stride (halves) for 32-half rows: banks col*20 mod 64 all distinct
#define KSTR    72   // LDS stride (halves) for 64-half rows: banks col*36 mod 64 all distinct
#define PSTR    40   // LDS stride (halves) for P tile rows

#if __has_builtin(__builtin_amdgcn_global_load_async_to_lds_b128)
#define HAVE_ASYNC_LDS 1
#endif

static __device__ __forceinline__ v8f wmma_f16(v16h a, v16h b, v8f c) {
  return __builtin_amdgcn_wmma_f32_16x16x32_f16(false, a, false, b, (short)0, c, false, false);
}

static __device__ __forceinline__ v16h cat8(v8h lo, v8h hi) {
  return __builtin_shufflevector(lo, hi, 0,1,2,3,4,5,6,7,8,9,10,11,12,13,14,15);
}

// 16-byte global->LDS copy: async on toolchains exposing the builtin.
static __device__ __forceinline__ void cp16(half_t* lds_dst, const half_t* gsrc) {
#if defined(HAVE_ASYNC_LDS)
  __builtin_amdgcn_global_load_async_to_lds_b128(
      (__attribute__((address_space(1))) v4i*)gsrc,
      (__attribute__((address_space(3))) v4i*)lds_dst, 0, 0);
#else
  *(v8h*)lds_dst = *(const v8h*)gsrc;
#endif
}

static __device__ __forceinline__ void wait_async_copies() {
#if defined(HAVE_ASYNC_LDS)
#if __has_builtin(__builtin_amdgcn_s_wait_asynccnt)
  __builtin_amdgcn_s_wait_asynccnt(0);
#else
  asm volatile("s_wait_asynccnt 0" ::: "memory");
#endif
#endif
}

// ---------------------------------------------------------------- prep kernels
__global__ void f32_to_f16_kernel(const float* __restrict__ in, half_t* __restrict__ out, int n) {
  int i = blockIdx.x * blockDim.x + threadIdx.x;
  if (i < n) out[i] = (half_t)in[i];
}

// in [rows][cols] f32 -> out [cols][rows] f16
__global__ void transpose_f16_kernel(const float* __restrict__ in, half_t* __restrict__ out,
                                     int rows, int cols) {
  int i = blockIdx.x * blockDim.x + threadIdx.x;
  if (i < rows * cols) {
    int r = i / cols, c = i - r * cols;
    out[c * rows + r] = (half_t)in[i];
  }
}

// ---------------------------------------------------------------- QKV GEMM
// Block = 8 waves: 128 rows x 64 cols. B panel (64 cols x 32 k = 4 KB) staged
// in double-buffered LDS by async copies, shared by all 8 waves (8x L2 saving).
__global__ __launch_bounds__(256) void qkv_gemm_kernel(
    const half_t* __restrict__ xh, const half_t* __restrict__ wT,
    const float* __restrict__ bqkv,
    half_t* __restrict__ Qb, half_t* __restrict__ Kb, half_t* __restrict__ Vtb) {
  __shared__ __attribute__((aligned(32))) half_t ldsB[2][64 * BSTR];  // 10 KB

  const int tid  = threadIdx.x;
  const int lane = tid & 31;
  const int wave = tid >> 5;
  const int NT   = (3 * HID) / 64;                 // 48 column panels
  const int bn   = blockIdx.x % NT;
  const int bm   = blockIdx.x / NT;                // [0,64)
  const int mBase = bm * 128 + wave * 16;
  const int nBase = bn * 64;
  const int col = lane & 15, hi = lane >> 4;

  // staging: 256 threads x 16 B; thread t -> panel row r, 8-half quarter q
  const int r = tid >> 2, q = tid & 3;
  const half_t* bsrc = wT + (size_t)(nBase + r) * HID + q * 8;   // + kk
  const int bDst = r * BSTR + q * 8;

  cp16(&ldsB[0][bDst], bsrc);
  wait_async_copies();
  __syncthreads();

  const half_t* arow = xh + (size_t)(mBase + col) * HID + hi * 16;
  v16h a = cat8(*(const v8h*)(arow), *(const v8h*)(arow + 8));

  v8f acc0 = {}, acc1 = {}, acc2 = {}, acc3 = {};
  int kb = 0;
  for (int kk = 0; kk < HID - 32; kk += 32, kb ^= 1) {
    cp16(&ldsB[kb ^ 1][bDst], bsrc + kk + 32);                    // stream next panel
    v16h an = cat8(*(const v8h*)(arow + kk + 32), *(const v8h*)(arow + kk + 40));
    const half_t* bp = &ldsB[kb][0] + col * BSTR + hi * 16;
    acc0 = wmma_f16(a, cat8(*(const v8h*)(bp),             *(const v8h*)(bp + 8)), acc0);
    acc1 = wmma_f16(a, cat8(*(const v8h*)(bp + 16 * BSTR), *(const v8h*)(bp + 16 * BSTR + 8)), acc1);
    acc2 = wmma_f16(a, cat8(*(const v8h*)(bp + 32 * BSTR), *(const v8h*)(bp + 32 * BSTR + 8)), acc2);
    acc3 = wmma_f16(a, cat8(*(const v8h*)(bp + 48 * BSTR), *(const v8h*)(bp + 48 * BSTR + 8)), acc3);
    a = an;
    __syncthreads();          // all waves done reading buffer kb
    wait_async_copies();      // next panel landed (this wave's copies)
    __syncthreads();          // whole next buffer valid
  }
  {
    const half_t* bp = &ldsB[kb][0] + col * BSTR + hi * 16;
    acc0 = wmma_f16(a, cat8(*(const v8h*)(bp),             *(const v8h*)(bp + 8)), acc0);
    acc1 = wmma_f16(a, cat8(*(const v8h*)(bp + 16 * BSTR), *(const v8h*)(bp + 16 * BSTR + 8)), acc1);
    acc2 = wmma_f16(a, cat8(*(const v8h*)(bp + 32 * BSTR), *(const v8h*)(bp + 32 * BSTR + 8)), acc2);
    acc3 = wmma_f16(a, cat8(*(const v8h*)(bp + 48 * BSTR), *(const v8h*)(bp + 48 * BSTR + 8)), acc3);
  }

  v8f accs[4] = {acc0, acc1, acc2, acc3};
  #pragma unroll
  for (int j = 0; j < 4; ++j) {
    const int o = nBase + j * 16 + col;            // qkv output column
    const float bq = bqkv[o];
    const int h = o / (3 * HD);                    // head
    const int rr = o - h * (3 * HD);               // 0..191 within head: q|k|v
    #pragma unroll
    for (int i = 0; i < 8; ++i) {
      const int t = mBase + i + 8 * hi;            // token
      const int b = t >> 10, s = t & (S_LEN - 1);
      const int bh = b * NH + h;
      const float v = accs[j][i] + bq;
      if (rr < HD) {
        Qb[((size_t)(bh * S_LEN + s)) * HD + rr] = (half_t)(v * 0.125f); // fold 1/sqrt(64)
      } else if (rr < 2 * HD) {
        Kb[((size_t)(bh * S_LEN + s)) * HD + (rr - HD)] = (half_t)v;
      } else {
        Vtb[((size_t)(bh * HD + (rr - 2 * HD))) * S_LEN + s] = (half_t)v;
      }
    }
  }
}

// ---------------------------------------------------------------- flash attention
// Block = 4 waves sharing one (b,h); each wave owns one 16-query tile.
__global__ __launch_bounds__(128) void attention_kernel(
    const half_t* __restrict__ Qb, const half_t* __restrict__ Kb,
    const half_t* __restrict__ Vtb, half_t* __restrict__ ctxh) {
  __shared__ __attribute__((aligned(32))) half_t ldsK[2][32 * KSTR];  // [buf][key][d+pad]
  __shared__ __attribute__((aligned(32))) half_t ldsV[2][HD * BSTR];  // [buf][d][key+pad]
  __shared__ __attribute__((aligned(32))) half_t Pt[4][16 * PSTR];    // per-wave P tile

  const int tid  = threadIdx.x;
  const int lane = tid & 31;
  const int wave = tid >> 5;
  const int bh   = blockIdx.x >> 4;                          // same for all 4 waves
  const int qt   = ((blockIdx.x & 15) << 2) + wave;          // query tile [0,64)
  const int col = lane & 15, hi = lane >> 4;

  // staging indices (128 threads): K rows 32x64h (4/row), V rows 64x32h (2/row)
  const int kr = tid >> 2, kq = tid & 3;                     // K: row, 16-half quarter
  const int vr = tid >> 1, vq = tid & 1;                     // V: row, 16-half half
  const half_t* kChunkBase = Kb  + (size_t)bh * S_LEN * HD;  // + k0*HD
  const half_t* vRowBase   = Vtb + (size_t)(bh * HD + vr) * S_LEN; // + k0
  const int kSrc = kr * HD + kq * 16;
  const int kDst = kr * KSTR + kq * 16;
  const int vDst = vr * BSTR + vq * 16;
  const int vSrc = vq * 16;

  // stage chunk 0 into buffer 0
  cp16(&ldsK[0][kDst],     kChunkBase + kSrc);
  cp16(&ldsK[0][kDst + 8], kChunkBase + kSrc + 8);
  cp16(&ldsV[0][vDst],     vRowBase + vSrc);
  cp16(&ldsV[0][vDst + 8], vRowBase + vSrc + 8);
  wait_async_copies();
  __syncthreads();

  // resident Q fragments (16 q x 64 d, pre-scaled by 1/8)
  const half_t* qrow = Qb + (size_t)(bh * S_LEN + qt * 16 + col) * HD + hi * 16;
  const v16h qa0 = cat8(*(const v8h*)(qrow),      *(const v8h*)(qrow + 8));  // d 0..31
  const v16h qa1 = cat8(*(const v8h*)(qrow + 32), *(const v8h*)(qrow + 40)); // d 32..63

  v8f O0 = {}, O1 = {}, O2 = {}, O3 = {};
  float m[8], l[8];
  #pragma unroll
  for (int i = 0; i < 8; ++i) { m[i] = -1e30f; l[i] = 0.0f; }

  half_t* Pw = &Pt[wave][0];
  const half_t* Pr = Pw + col * PSTR + hi * 16;    // A-fragment read pattern

  int kb = 0;
  for (int k0 = 0; k0 < S_LEN; k0 += 32, kb ^= 1) {
    if (k0 + 32 < S_LEN) {                                   // stream next chunk
      const half_t* kn = kChunkBase + (k0 + 32) * HD;
      const half_t* vn = vRowBase + (k0 + 32);
      cp16(&ldsK[kb ^ 1][kDst],     kn + kSrc);
      cp16(&ldsK[kb ^ 1][kDst + 8], kn + kSrc + 8);
      cp16(&ldsV[kb ^ 1][vDst],     vn + vSrc);
      cp16(&ldsV[kb ^ 1][vDst + 8], vn + vSrc + 8);
    }

    // scores: S(16x32) = Q(16x64) @ K_chunk^T  (K chunk in LDS, [key][d])
    const half_t* kcol = &ldsK[kb][0] + col * KSTR + hi * 16;
    v8f s0 = {}, s1 = {};
    s0 = wmma_f16(qa0, cat8(*(const v8h*)(kcol),      *(const v8h*)(kcol + 8)),  s0);
    s0 = wmma_f16(qa1, cat8(*(const v8h*)(kcol + 32), *(const v8h*)(kcol + 40)), s0);
    s1 = wmma_f16(qa0, cat8(*(const v8h*)(kcol + 16 * KSTR),      *(const v8h*)(kcol + 16 * KSTR + 8)),  s1);
    s1 = wmma_f16(qa1, cat8(*(const v8h*)(kcol + 16 * KSTR + 32), *(const v8h*)(kcol + 16 * KSTR + 40)), s1);

    // load V fragments early so ds latency hides under the softmax VALU work
    const half_t* vcol = &ldsV[kb][0] + col * BSTR + hi * 16;
    const v16h vb0 = cat8(*(const v8h*)(vcol),             *(const v8h*)(vcol + 8));
    const v16h vb1 = cat8(*(const v8h*)(vcol + 16 * BSTR), *(const v8h*)(vcol + 16 * BSTR + 8));
    const v16h vb2 = cat8(*(const v8h*)(vcol + 32 * BSTR), *(const v8h*)(vcol + 32 * BSTR + 8));
    const v16h vb3 = cat8(*(const v8h*)(vcol + 48 * BSTR), *(const v8h*)(vcol + 48 * BSTR + 8));

    // online softmax; each row M lives on 16 lanes of one half-wave -> xor 1/2/4/8
    #pragma unroll
    for (int i = 0; i < 8; ++i) {
      float t = fmaxf(s0[i], s1[i]);
      t = fmaxf(t, __shfl_xor(t, 1));
      t = fmaxf(t, __shfl_xor(t, 2));
      t = fmaxf(t, __shfl_xor(t, 4));
      t = fmaxf(t, __shfl_xor(t, 8));
      const float mn = fmaxf(m[i], t);
      const float corr = __expf(m[i] - mn);
      m[i] = mn;
      const float p0 = __expf(s0[i] - mn);
      const float p1 = __expf(s1[i] - mn);
      float rs = p0 + p1;
      rs += __shfl_xor(rs, 1);
      rs += __shfl_xor(rs, 2);
      rs += __shfl_xor(rs, 4);
      rs += __shfl_xor(rs, 8);
      l[i] = l[i] * corr + rs;
      O0[i] *= corr; O1[i] *= corr; O2[i] *= corr; O3[i] *= corr;
      Pw[(i + 8 * hi) * PSTR + col]      = (half_t)p0;
      Pw[(i + 8 * hi) * PSTR + 16 + col] = (half_t)p1;
    }
    asm volatile("s_wait_dscnt 0" ::: "memory");

    // P(16x32) as A-fragment from LDS; V chunk [d][key] gives col-major B
    const v16h pa = cat8(*(const v8h*)(Pr), *(const v8h*)(Pr + 8));
    O0 = wmma_f16(pa, vb0, O0);
    O1 = wmma_f16(pa, vb1, O1);
    O2 = wmma_f16(pa, vb2, O2);
    O3 = wmma_f16(pa, vb3, O3);

    __syncthreads();          // all waves done reading buffer kb
    wait_async_copies();      // this wave's copies into kb^1 have landed
    __syncthreads();          // whole next buffer valid for everyone
  }

  // normalize and write ctx [token][h*64+d] in f16
  const int b = bh >> 4, h = bh & (NH - 1);
  #pragma unroll
  for (int i = 0; i < 8; ++i) {
    const int s = qt * 16 + i + 8 * hi;
    const float inv = 1.0f / l[i];
    const size_t base = ((size_t)(b * S_LEN + s)) * HID + h * HD;
    ctxh[base +  0 + col] = (half_t)(O0[i] * inv);
    ctxh[base + 16 + col] = (half_t)(O1[i] * inv);
    ctxh[base + 32 + col] = (half_t)(O2[i] * inv);
    ctxh[base + 48 + col] = (half_t)(O3[i] * inv);
  }
}

// ---------------------------------------------------------------- output projection
__global__ __launch_bounds__(256) void out_gemm_kernel(
    const half_t* __restrict__ ctxh, const half_t* __restrict__ wT,
    const float* __restrict__ bout, float* __restrict__ out) {
  __shared__ __attribute__((aligned(32))) half_t ldsB[2][64 * BSTR];  // 10 KB

  const int tid  = threadIdx.x;
  const int lane = tid & 31;
  const int wave = tid >> 5;
  const int NT   = HID / 64;                       // 16 column panels
  const int bn   = blockIdx.x % NT;
  const int bm   = blockIdx.x / NT;                // [0,64)
  const int mBase = bm * 128 + wave * 16;
  const int nBase = bn * 64;
  const int col = lane & 15, hi = lane >> 4;

  const int r = tid >> 2, q = tid & 3;
  const half_t* bsrc = wT + (size_t)(nBase + r) * HID + q * 8;
  const int bDst = r * BSTR + q * 8;

  cp16(&ldsB[0][bDst], bsrc);
  wait_async_copies();
  __syncthreads();

  const half_t* arow = ctxh + (size_t)(mBase + col) * HID + hi * 16;
  v16h a = cat8(*(const v8h*)(arow), *(const v8h*)(arow + 8));

  v8f acc0 = {}, acc1 = {}, acc2 = {}, acc3 = {};
  int kb = 0;
  for (int kk = 0; kk < HID - 32; kk += 32, kb ^= 1) {
    cp16(&ldsB[kb ^ 1][bDst], bsrc + kk + 32);
    v16h an = cat8(*(const v8h*)(arow + kk + 32), *(const v8h*)(arow + kk + 40));
    const half_t* bp = &ldsB[kb][0] + col * BSTR + hi * 16;
    acc0 = wmma_f16(a, cat8(*(const v8h*)(bp),             *(const v8h*)(bp + 8)), acc0);
    acc1 = wmma_f16(a, cat8(*(const v8h*)(bp + 16 * BSTR), *(const v8h*)(bp + 16 * BSTR + 8)), acc1);
    acc2 = wmma_f16(a, cat8(*(const v8h*)(bp + 32 * BSTR), *(const v8h*)(bp + 32 * BSTR + 8)), acc2);
    acc3 = wmma_f16(a, cat8(*(const v8h*)(bp + 48 * BSTR), *(const v8h*)(bp + 48 * BSTR + 8)), acc3);
    a = an;
    __syncthreads();
    wait_async_copies();
    __syncthreads();
  }
  {
    const half_t* bp = &ldsB[kb][0] + col * BSTR + hi * 16;
    acc0 = wmma_f16(a, cat8(*(const v8h*)(bp),             *(const v8h*)(bp + 8)), acc0);
    acc1 = wmma_f16(a, cat8(*(const v8h*)(bp + 16 * BSTR), *(const v8h*)(bp + 16 * BSTR + 8)), acc1);
    acc2 = wmma_f16(a, cat8(*(const v8h*)(bp + 32 * BSTR), *(const v8h*)(bp + 32 * BSTR + 8)), acc2);
    acc3 = wmma_f16(a, cat8(*(const v8h*)(bp + 48 * BSTR), *(const v8h*)(bp + 48 * BSTR + 8)), acc3);
  }

  v8f accs[4] = {acc0, acc1, acc2, acc3};
  #pragma unroll
  for (int j = 0; j < 4; ++j) {
    const int o = nBase + j * 16 + col;
    const float bo = bout[o];
    #pragma unroll
    for (int i = 0; i < 8; ++i) {
      const int t = mBase + i + 8 * hi;
      out[(size_t)t * HID + o] = accs[j][i] + bo;
    }
  }
}

// ---------------------------------------------------------------- launch
extern "C" void kernel_launch(void* const* d_in, const int* in_sizes, int n_in,
                              void* d_out, int out_size, void* d_ws, size_t ws_size,
                              hipStream_t stream) {
  (void)in_sizes; (void)n_in; (void)out_size; (void)ws_size;
  const float* x      = (const float*)d_in[0];   // [8,1024,1024]
  const float* w_qkv  = (const float*)d_in[1];   // [1024,3072]
  const float* b_qkv  = (const float*)d_in[2];   // [3072]
  const float* w_out  = (const float*)d_in[3];   // [1024,1024]
  const float* b_out  = (const float*)d_in[4];   // [1024]
  float* out = (float*)d_out;                    // [8,1024,1024]

  char* ws = (char*)d_ws;
  size_t off = 0;
  auto take = [&](size_t elems) { half_t* p = (half_t*)(ws + off); off += elems * sizeof(half_t); return p; };
  half_t* xh    = take((size_t)TOKENS * HID);            // 16 MB
  half_t* wqkvT = take((size_t)3 * HID * HID);           //  6 MB
  half_t* woutT = take((size_t)HID * HID);               //  2 MB
  half_t* Qb    = take((size_t)BATCH * NH * S_LEN * HD); // 16 MB
  half_t* Kb    = take((size_t)BATCH * NH * S_LEN * HD); // 16 MB
  half_t* Vtb   = take((size_t)BATCH * NH * HD * S_LEN); // 16 MB
  half_t* ctxh  = take((size_t)TOKENS * HID);            // 16 MB  (total 88 MB)

  const int nx = TOKENS * HID;                   // 8388608
  f32_to_f16_kernel<<<nx / 256, 256, 0, stream>>>(x, xh, nx);
  transpose_f16_kernel<<<(HID * 3 * HID + 255) / 256, 256, 0, stream>>>(w_qkv, wqkvT, HID, 3 * HID);
  transpose_f16_kernel<<<(HID * HID + 255) / 256, 256, 0, stream>>>(w_out, woutT, HID, HID);

  qkv_gemm_kernel<<<3072, 256, 0, stream>>>(xh, wqkvT, b_qkv, Qb, Kb, Vtb);
  attention_kernel<<<2048, 128, 0, stream>>>(Qb, Kb, Vtb, ctxh);
  out_gemm_kernel<<<1024, 256, 0, stream>>>(ctxh, woutT, b_out, out);
}
